// ConvBranch_14224931684837
// MI455X (gfx1250) — matrix operand order repaired and checked by reading, not compiled, not run on hardware
//
#include <hip/hip_runtime.h>
#include <math.h>
#include <stdint.h>

// ---------------- problem constants ----------------
#define Bn 4
#define Tn 2048
#define Wn 1024
#define Kt 7
#define Gn 8
#define Cgn 128
#define NT 8            // N tiles per group (128/16)
#define RSTEPS 28       // 896 / 32
#define KC 896          // Kt*Cgn reduction length per group
#define AROW 904        // padded LDS row stride (bf16 elems): 1808B -> 452 dwords, 452%64=4
#define LN_EPS 1e-5f
#define TCH 16          // T chunks for column-sum parallelism
#define TPC (Tn / TCH)  // 128 rows per chunk

// ---------------- workspace layout (bytes) ----------------
#define OFF_OFFS   0u                  // [B*T*K] f32            = 229376
#define OFF_MU     229376u             // [B*T] f32              = 32768
#define OFF_RSTD   262144u             // [B*T] f32              = 32768
#define OFF_WPACK  294912u             // [8*8*28*32*16] bf16    = 1835008
#define OFF_XCONV  2129920u            // [B*T*W] f32            = 33554432
#define OFF_S      35684352u           // [B*W] f32              = 16384
#define OFF_PART   35700736u           // [B*TCH*W] f32          = 262144
#define OFF_LEN    35962880u           // [B] f32                = 16

typedef __attribute__((ext_vector_type(16))) __bf16        v16bf;
typedef __attribute__((ext_vector_type(8)))  float         v8f;
typedef __attribute__((ext_vector_type(4)))  unsigned int  u32x4;

union FragU { u32x4 u[2]; v16bf v; };

static __device__ __forceinline__ unsigned short f2bf(float f) {
    unsigned int u = __float_as_uint(f);
    unsigned int r = 0x7fffu + ((u >> 16) & 1u);   // round-to-nearest-even
    return (unsigned short)((u + r) >> 16);
}

// ================= kernel 1: offsets = h @ offset_w + offset_b =================
// one wave (32 lanes) per (b,t) row; offset_w staged in LDS
__global__ void k_offsets(const float* __restrict__ h,
                          const float* __restrict__ offset_w,
                          const float* __restrict__ offset_b,
                          float* __restrict__ offs) {
    __shared__ float ow_sh[Wn * Kt];          // 28672 B
    const int tid  = threadIdx.x;
    for (int i = tid; i < Wn * Kt; i += 256) ow_sh[i] = offset_w[i];
    __syncthreads();

    const int lane = tid & 31;
    const int wv   = tid >> 5;
    const int row  = blockIdx.x * 8 + wv;     // 1024 blocks * 8 = 8192 rows

    float acc[Kt];
#pragma unroll
    for (int k = 0; k < Kt; ++k) acc[k] = 0.f;

    const float* hrow = h + (size_t)row * Wn;
    for (int i = 0; i < Wn / 32; ++i) {
        int w = i * 32 + lane;
        float hv = hrow[w];
#pragma unroll
        for (int k = 0; k < Kt; ++k) acc[k] += hv * ow_sh[w * Kt + k];
    }
#pragma unroll
    for (int k = 0; k < Kt; ++k)
        for (int off = 16; off > 0; off >>= 1)
            acc[k] += __shfl_xor(acc[k], off, 32);

    if (lane == 0) {
#pragma unroll
        for (int k = 0; k < Kt; ++k)
            offs[row * Kt + k] = acc[k] + offset_b[k];
    }
}

// ================= kernel 2: pack conv_w into WMMA B-fragment layout =================
// wpack[(((g*NT+nt)*RSTEPS+r)*32+lane)*16+e] = bf16(conv_w[g][o][c][ktap])
//   o = nt*16 + (lane&15); kc = r*32 + (lane>>4)*16 + e; ktap = kc>>7; c = kc&127
__global__ void k_pack(const float* __restrict__ conv_w,
                       unsigned short* __restrict__ wpack) {
    int i = blockIdx.x * 256 + threadIdx.x;    // 3584 blocks -> 917504 elems
    int e     = i & 15;
    int lane  = (i >> 4) & 31;
    int rest  = i >> 9;
    int r     = rest % RSTEPS;
    int rest2 = rest / RSTEPS;
    int nt    = rest2 & 7;
    int g     = rest2 >> 3;

    int kc   = r * 32 + (lane >> 4) * 16 + e;
    int ktap = kc >> 7;
    int c    = kc & 127;
    int o    = nt * 16 + (lane & 15);
    float v  = conv_w[(((size_t)g * Cgn + o) * Cgn + c) * Kt + ktap];
    wpack[i] = f2bf(v);
}

// ================= kernel 3: deformable gather + grouped-conv GEMM (WMMA bf16) =================
// grid.x = 512 M-tiles (16 rows), grid.y = 8 groups; 8 waves, wave = one 16-ch N tile
__global__ void k_conv(const float* __restrict__ h,
                       const float* __restrict__ offs,
                       const unsigned short* __restrict__ wpack,
                       const float* __restrict__ conv_b,
                       float* __restrict__ xconv) {
    __shared__ __align__(16) unsigned short Ash[16 * AROW];   // 28928 B
    __shared__ int   p0s[16 * Kt];
    __shared__ int   p1s[16 * Kt];
    __shared__ float frs[16 * Kt];

    const int tid  = threadIdx.x;
    const int tile = blockIdx.x;
    const int g    = blockIdx.y;
    const int row0 = tile * 16;
    const int b    = row0 / Tn;                // 16-row tile never spans batch boundary
    const int gch  = g * Cgn;

    // ---- per (m, tap) sampling positions ----
    if (tid < 16 * Kt) {
        int m = tid / Kt, k = tid % Kt;
        int row = row0 + m;
        int t   = row - b * Tn;
        float off  = offs[row * Kt + k];
        float posf = (float)t + (float)(k - (Kt - 1) / 2) + off;
        posf = fminf(fmaxf(posf, 0.f), (float)(Tn - 1));
        int p0 = (int)floorf(posf);
        int p1 = p0 + 1; if (p1 > Tn - 1) p1 = Tn - 1;
        p0s[tid] = p0; p1s[tid] = p1; frs[tid] = posf - (float)p0;
    }
    __syncthreads();

    // ---- build A tile: interpolated samples, bf16, tap-major kc = k*128 + c ----
    for (int idx = tid; idx < 16 * KC; idx += 256) {
        int m  = idx / KC;
        int kc = idx - m * KC;
        int k  = kc >> 7;
        int c  = kc & 127;
        int pi = m * Kt + k;
        int p0 = p0s[pi], p1 = p1s[pi];
        float fr = frs[pi];
        float v0 = h[((size_t)(b * Tn + p0)) * Wn + gch + c];
        float v1 = h[((size_t)(b * Tn + p1)) * Wn + gch + c];
        Ash[m * AROW + kc] = f2bf(v0 + fr * (v1 - v0));
    }
    __syncthreads();

    // ---- WMMA main loop: wave wv owns output channels [g*128 + wv*16 .. +15] ----
    const int lane = tid & 31;
    const int wv   = tid >> 5;
    const int half = lane >> 4;
    const int mr   = lane & 15;

    const unsigned short* wbase = wpack + ((size_t)(g * NT + wv) * RSTEPS) * 32 * 16;

    v8f acc = {};
    for (int r = 0; r < RSTEPS; ++r) {
        // A fragment: lane holds M=mr; K contiguous r*32+half*16 .. +15
        const u32x4* ap = (const u32x4*)&Ash[mr * AROW + r * 32 + half * 16];
        FragU a; a.u[0] = ap[0]; a.u[1] = ap[1];
        // B fragment: contiguous 32B per lane from pre-swizzled weights
        const u32x4* bp = (const u32x4*)(wbase + ((size_t)(r * 32 + lane)) * 16);
        FragU bb; bb.u[0] = bp[0]; bb.u[1] = bp[1];
        acc = __builtin_amdgcn_wmma_f32_16x16x32_bf16(
            false, a.v, false, bb.v, (short)0, acc, false, false);
    }

    // ---- epilogue: lane l -> N = l&15, VGPR v -> M = v + 8*(l>>4) ----
    const int o    = g * Cgn + wv * 16 + mr;
    const float bias = conv_b[o];
#pragma unroll
    for (int v = 0; v < 8; ++v) {
        int m = half * 8 + v;
        xconv[((size_t)(row0 + m)) * Wn + o] = acc[v] + bias;
    }
}

// ================= kernel 4: per-row LN statistics =================
__global__ void k_lnstats(const float* __restrict__ xconv,
                          float* __restrict__ mu, float* __restrict__ rstd) {
    __shared__ float redS[8], redQ[8];
    const int tid = threadIdx.x, lane = tid & 31, wv = tid >> 5;
    const int row = blockIdx.x;
    const float* xr = xconv + (size_t)row * Wn;
    float s = 0.f, q = 0.f;
#pragma unroll
    for (int i = 0; i < 4; ++i) {
        float x = xr[tid + 256 * i];
        s += x; q += x * x;
    }
    for (int off = 16; off > 0; off >>= 1) {
        s += __shfl_xor(s, off, 32);
        q += __shfl_xor(q, off, 32);
    }
    if (lane == 0) { redS[wv] = s; redQ[wv] = q; }
    __syncthreads();
    if (tid == 0) {
        float S = 0.f, Q = 0.f;
        for (int i = 0; i < 8; ++i) { S += redS[i]; Q += redQ[i]; }
        float m = S / (float)Wn;
        float var = Q / (float)Wn - m * m;
        mu[row]   = m;
        rstd[row] = rsqrtf(var + LN_EPS);
    }
}

// ================= kernel 5a: masked valid lengths per batch =================
__global__ void k_len(const unsigned char* __restrict__ mask,
                      float* __restrict__ len) {
    __shared__ float red[256];
    const int tid = threadIdx.x;
    const int b   = blockIdx.x;
    float c = 0.f;
    for (int t = tid; t < Tn; t += 256) c += mask[b * Tn + t] ? 0.f : 1.f;
    red[tid] = c;
    __syncthreads();
    for (int off = 128; off > 0; off >>= 1) {
        if (tid < off) red[tid] += red[tid + off];
        __syncthreads();
    }
    if (tid == 0) len[b] = fmaxf(red[0], 1.f);
}

// ================= kernel 5b: partial masked column sums (LN applied on the fly) =================
// grid = (W/256, B, TCH); each block sums TPC rows of one T-chunk
__global__ void k_colsum_part(const float* __restrict__ xconv,
                              const unsigned char* __restrict__ mask,
                              const float* __restrict__ mu, const float* __restrict__ rstd,
                              float* __restrict__ part) {
    const int b  = blockIdx.y;
    const int tz = blockIdx.z;
    const int w  = blockIdx.x * 256 + threadIdx.x;
    const int t0 = tz * TPC;
    float acc = 0.f;
    for (int t = t0; t < t0 + TPC; ++t) {
        int row = b * Tn + t;
        float valid = mask[row] ? 0.f : 1.f;
        float x = xconv[(size_t)row * Wn + w];
        acc += valid * (x - mu[row]) * rstd[row];
    }
    part[((size_t)(b * TCH + tz)) * Wn + w] = acc;
}

// ================= kernel 5c: finalize s[b][w] = ln_g*acc/len + ln_b =================
__global__ void k_colsum_fin(const float* __restrict__ part,
                             const float* __restrict__ len,
                             const float* __restrict__ ln_g, const float* __restrict__ ln_b,
                             float* __restrict__ s) {
    const int b = blockIdx.y;
    const int w = blockIdx.x * 256 + threadIdx.x;
    float acc = 0.f;
#pragma unroll
    for (int tz = 0; tz < TCH; ++tz)
        acc += part[((size_t)(b * TCH + tz)) * Wn + w];
    s[b * Wn + w] = ln_g[w] * (acc / len[b]) + ln_b[w];
}

// ================= kernel 6: SE MLP + gate + final projection -> out[B] =================
__global__ void k_se_out(const float* __restrict__ s,
                         const float* __restrict__ se_w1, const float* __restrict__ se_b1,
                         const float* __restrict__ se_w2, const float* __restrict__ se_b2,
                         const float* __restrict__ proj_w, const float* __restrict__ proj_b,
                         float* __restrict__ out) {
    __shared__ float s_sh[Bn * Wn];    // 16 KB
    __shared__ float hid[Bn * 64];
    __shared__ float red[256];
    const int tid = threadIdx.x;
    for (int i = tid; i < Bn * Wn; i += 256) s_sh[i] = s[i];
    __syncthreads();

    // hidden = relu(s @ w1 + b1): thread (b,j), b=tid/64, j=tid%64
    {
        int b = tid >> 6, j = tid & 63;
        float a = 0.f;
        for (int w = 0; w < Wn; ++w) a += s_sh[b * Wn + w] * se_w1[w * 64 + j];
        a += se_b1[j];
        hid[b * 64 + j] = a > 0.f ? a : 0.f;
    }
    __syncthreads();

    // out[b] = proj_b + sum_w sigmoid(hid@w2 + b2)[w] * s[b][w] * proj_w[w]
    {
        int b = tid >> 6, j0 = tid & 63;
        float part = 0.f;
        for (int w = j0; w < Wn; w += 64) {
            float a = se_b2[w];
            for (int j = 0; j < 64; ++j) a += hid[b * 64 + j] * se_w2[j * Wn + w];
            float gate = 1.f / (1.f + expf(-a));
            part += gate * s_sh[b * Wn + w] * proj_w[w];
        }
        red[tid] = part;
    }
    __syncthreads();
    for (int off = 32; off > 0; off >>= 1) {
        if ((tid & 63) < off) red[tid] += red[tid + off];
        __syncthreads();
    }
    if ((tid & 63) == 0) out[tid >> 6] = red[tid] + proj_b[0];
}

// ================= host launcher =================
extern "C" void kernel_launch(void* const* d_in, const int* in_sizes, int n_in,
                              void* d_out, int out_size, void* d_ws, size_t ws_size,
                              hipStream_t stream) {
    const float*         h        = (const float*)d_in[0];
    const unsigned char* mask     = (const unsigned char*)d_in[1];   // numpy bool = 1 byte
    const float*         offset_w = (const float*)d_in[2];
    const float*         offset_b = (const float*)d_in[3];
    const float*         conv_w   = (const float*)d_in[4];
    const float*         conv_b   = (const float*)d_in[5];
    const float*         ln_g     = (const float*)d_in[6];
    const float*         ln_b     = (const float*)d_in[7];
    const float*         se_w1    = (const float*)d_in[8];
    const float*         se_b1    = (const float*)d_in[9];
    const float*         se_w2    = (const float*)d_in[10];
    const float*         se_b2    = (const float*)d_in[11];
    const float*         proj_w   = (const float*)d_in[12];
    const float*         proj_b   = (const float*)d_in[13];
    float* out = (float*)d_out;

    char* ws = (char*)d_ws;
    float*          offs  = (float*)(ws + OFF_OFFS);
    float*          mu    = (float*)(ws + OFF_MU);
    float*          rstd  = (float*)(ws + OFF_RSTD);
    unsigned short* wpack = (unsigned short*)(ws + OFF_WPACK);
    float*          xconv = (float*)(ws + OFF_XCONV);
    float*          sbuf  = (float*)(ws + OFF_S);
    float*          part  = (float*)(ws + OFF_PART);
    float*          len   = (float*)(ws + OFF_LEN);

    k_offsets    <<<dim3(Bn * Tn / 8), dim3(256), 0, stream>>>(h, offset_w, offset_b, offs);
    k_pack       <<<dim3(Gn * NT * RSTEPS * 32 * 16 / 256), dim3(256), 0, stream>>>(conv_w, wpack);
    k_conv       <<<dim3(Bn * Tn / 16, Gn), dim3(256), 0, stream>>>(h, offs, wpack, conv_b, xconv);
    k_lnstats    <<<dim3(Bn * Tn), dim3(256), 0, stream>>>(xconv, mu, rstd);
    k_len        <<<dim3(Bn), dim3(256), 0, stream>>>(mask, len);
    k_colsum_part<<<dim3(Wn / 256, Bn, TCH), dim3(256), 0, stream>>>(xconv, mask, mu, rstd, part);
    k_colsum_fin <<<dim3(Wn / 256, Bn), dim3(256), 0, stream>>>(part, len, ln_g, ln_b, sbuf);
    k_se_out     <<<dim3(1), dim3(256), 0, stream>>>(sbuf, se_w1, se_b1, se_w2, se_b2, proj_w, proj_b, out);
}
